// CustomNetwork_EdgeAttentionWithNodeEncoding_54193897341339
// MI455X (gfx1250) — compile-verified, matrix operand order checked by reading, not run on hardware
//
#include <hip/hip_runtime.h>
#include <hip/hip_bf16.h>

typedef __attribute__((ext_vector_type(16))) __bf16 v16bf;
typedef __attribute__((ext_vector_type(8)))  float  v8f;

constexpr int kB   = 2;
constexpr int kM   = 48;
constexpr int kD   = 128;
constexpr int kH   = 4;
constexpr int kL   = kM * kM;     // 2304
constexpr int kDin = 3 * kD;      // 384
constexpr int kHd  = kDin / kH;   // 96 (big-MHA head dim)

// ---------------------------------------------------------------------------
// helpers
// ---------------------------------------------------------------------------
__device__ __forceinline__ v8f wmma_bf16(v16bf a, v16bf b, v8f c) {
  // (neg_a, A, neg_b, B, c_mod, C, reuse_a, reuse_b)
  return __builtin_amdgcn_wmma_f32_16x16x32_bf16(false, a, false, b, (short)0, c,
                                                 false, false);
}

// K index held by fragment element e of `lane` (16-bit A/B layout, 16x32)
__device__ __forceinline__ int kmap(int e, int lane) {
  return (e & 7) + ((e >> 3) << 4) + ((lane >> 4) << 3);
}

// ---------------------------------------------------------------------------
// conversion / transpose kernels (run once per produced buffer / weight)
// ---------------------------------------------------------------------------
__global__ void f32_to_bf16_kernel(const float* __restrict__ src,
                                   __bf16* __restrict__ dst, int n) {
  int i = blockIdx.x * 256 + threadIdx.x;
  if (i < n) dst[i] = (__bf16)src[i];
}

// W (Kin,Kout) fp32 row-major  ->  WT (Kout,Kin) bf16 row-major
__global__ void transpose_to_bf16_kernel(const float* __restrict__ W,
                                         __bf16* __restrict__ WT,
                                         int Kin, int Kout) {
  int idx = blockIdx.x * 256 + threadIdx.x;
  if (idx >= Kin * Kout) return;
  int k = idx / Kout, n = idx % Kout;
  WT[(size_t)n * Kin + k] = (__bf16)W[idx];
}

// ---------------------------------------------------------------------------
// GEMM, 16x16 tile per wave (small N): C[N,Kout] = A[N,Kin] @ WT[Kout,Kin]^T
// All operands bf16, fp32 accumulate. Both fragments are contiguous loads.
// ---------------------------------------------------------------------------
__global__ void wmma_gemm_n16(const __bf16* __restrict__ A,
                              const __bf16* __restrict__ WT,
                              __bf16* __restrict__ C, int Kin, int Kout) {
  const int lane = threadIdx.x;
  const int col  = lane & 15;
  const int kb   = (lane >> 4) << 3;
  const int m0   = blockIdx.y << 4;
  const int n0   = blockIdx.x << 4;
  v8f acc = {};
  const __bf16* arow = A + (size_t)(m0 + col) * Kin;
  const __bf16* brow = WT + (size_t)(n0 + col) * Kin;
  for (int kt = 0; kt < Kin; kt += 32) {
    v16bf a, b;
    const __bf16* ap = arow + kt + kb;
    const __bf16* bp = brow + kt + kb;
#pragma unroll
    for (int e = 0; e < 8; ++e) {
      a[e] = ap[e]; a[e + 8] = ap[16 + e];
      b[e] = bp[e]; b[e + 8] = bp[16 + e];
    }
    acc = wmma_bf16(a, b, acc);
  }
  const int rb = (lane >> 4) << 3;
#pragma unroll
  for (int r = 0; r < 8; ++r)
    C[(size_t)(m0 + rb + r) * Kout + n0 + col] = (__bf16)acc[r];
}

// ---------------------------------------------------------------------------
// GEMM, 64x16 tile per wave: B fragment amortized over 4 row tiles; 4
// independent WMMAs per k-step (no D->A/B RAW hazard stalls). N % 64 == 0.
// ---------------------------------------------------------------------------
__global__ void wmma_gemm_m64(const __bf16* __restrict__ A,
                              const __bf16* __restrict__ WT,
                              __bf16* __restrict__ C, int Kin, int Kout) {
  const int lane = threadIdx.x;
  const int col  = lane & 15;
  const int kb   = (lane >> 4) << 3;
  const int m0   = blockIdx.y << 6;
  const int n0   = blockIdx.x << 4;
  v8f zero = {};
  v8f acc[4];
#pragma unroll
  for (int mt = 0; mt < 4; ++mt) acc[mt] = zero;

  const __bf16* brow = WT + (size_t)(n0 + col) * Kin;
  for (int kt = 0; kt < Kin; kt += 32) {
    v16bf b;
    const __bf16* bp = brow + kt + kb;
#pragma unroll
    for (int e = 0; e < 8; ++e) { b[e] = bp[e]; b[e + 8] = bp[16 + e]; }
    __builtin_prefetch(bp + 32, 0, 0);   // global_prefetch_b8
#pragma unroll
    for (int mt = 0; mt < 4; ++mt) {
      const __bf16* ap = A + (size_t)(m0 + mt * 16 + col) * Kin + kt + kb;
      v16bf a;
#pragma unroll
      for (int e = 0; e < 8; ++e) { a[e] = ap[e]; a[e + 8] = ap[16 + e]; }
      acc[mt] = wmma_bf16(a, b, acc[mt]);
    }
  }
  const int rb = (lane >> 4) << 3;
#pragma unroll
  for (int mt = 0; mt < 4; ++mt)
#pragma unroll
    for (int r = 0; r < 8; ++r)
      C[(size_t)(m0 + mt * 16 + rb + r) * Kout + n0 + col] = (__bf16)acc[mt][r];
}

// ---------------------------------------------------------------------------
// Flash-style masked MHA over L=2304, head dim 96, key-mask only.
// grid = (L/16, H, B), block = 32. Q,K,V,O: (B,L,384) bf16. km: (B,L) fp32.
// ---------------------------------------------------------------------------
__global__ void mha_attn_kernel(const __bf16* __restrict__ Q,
                                const __bf16* __restrict__ K,
                                const __bf16* __restrict__ V,
                                const float* __restrict__ km,
                                __bf16* __restrict__ O) {
  const int lane = threadIdx.x;
  const int col  = lane & 15;
  const int kb   = (lane >> 4) << 3;
  const int q0   = blockIdx.x << 4;
  const int h    = blockIdx.y;
  const int b    = blockIdx.z;
  const float scale = 0.10206207261596577f; // 1/sqrt(96)
  __shared__ __bf16 pl[16 * 32];

  // Q tile fragments (16 x 96 -> three 16x32 A fragments), contiguous bf16
  v16bf aQ[3];
  const __bf16* qrow = Q + ((size_t)b * kL + q0 + col) * kDin + h * kHd;
#pragma unroll
  for (int dt = 0; dt < 3; ++dt) {
    const __bf16* p = qrow + dt * 32 + kb;
#pragma unroll
    for (int e = 0; e < 8; ++e) { aQ[dt][e] = p[e]; aQ[dt][e + 8] = p[16 + e]; }
  }

  v8f zero = {};
  v8f accO[6];
#pragma unroll
  for (int t = 0; t < 6; ++t) accO[t] = zero;
  float m8[8], l8[8];
#pragma unroll
  for (int r = 0; r < 8; ++r) { m8[r] = -1e30f; l8[r] = 0.f; }

  for (int k0 = 0; k0 < kL; k0 += 32) {
    // --- scores: S(16x32) = Q(16x96) * K^T ---
    v8f s0 = zero, s1 = zero;
    const __bf16* kr0 = K + ((size_t)b * kL + k0 + col) * kDin + h * kHd;
    const __bf16* kr1 = kr0 + (size_t)16 * kDin;
#pragma unroll
    for (int dt = 0; dt < 3; ++dt) {
      v16bf b0, b1;
      const __bf16* p0 = kr0 + dt * 32 + kb;
      const __bf16* p1 = kr1 + dt * 32 + kb;
#pragma unroll
      for (int e = 0; e < 8; ++e) {
        b0[e] = p0[e]; b0[e + 8] = p0[16 + e];
        b1[e] = p1[e]; b1[e + 8] = p1[16 + e];
      }
      s0 = wmma_bf16(aQ[dt], b0, s0);
      s1 = wmma_bf16(aQ[dt], b1, s1);
    }
    const float km0 = km[(size_t)b * kL + k0 + col];
    const float km1 = km[(size_t)b * kL + k0 + 16 + col];

    // --- online softmax update (rows live on fixed 16-lane halves) ---
    float p0a[8], p1a[8];
#pragma unroll
    for (int r = 0; r < 8; ++r) {
      float v0 = (km0 == 0.f) ? -1e9f : s0[r] * scale;
      float v1 = (km1 == 0.f) ? -1e9f : s1[r] * scale;
      float mx = fmaxf(v0, v1);
#pragma unroll
      for (int off = 1; off < 16; off <<= 1) mx = fmaxf(mx, __shfl_xor(mx, off));
      float mn = fmaxf(m8[r], mx);
      float sf = __expf(m8[r] - mn);
      m8[r] = mn;
      float e0 = __expf(v0 - mn), e1 = __expf(v1 - mn);
      float rs = e0 + e1;
#pragma unroll
      for (int off = 1; off < 16; off <<= 1) rs += __shfl_xor(rs, off);
      l8[r] = l8[r] * sf + rs;
#pragma unroll
      for (int t = 0; t < 6; ++t) accO[t][r] *= sf;
      p0a[r] = e0; p1a[r] = e1;
    }

    // --- C-layout P -> LDS (bf16) -> A-fragment layout ---
    const int prow = (lane >> 4) << 3;
#pragma unroll
    for (int r = 0; r < 8; ++r) {
      pl[(prow + r) * 32 + col]      = (__bf16)p0a[r];
      pl[(prow + r) * 32 + 16 + col] = (__bf16)p1a[r];
    }
    __syncthreads();
    v16bf aP;
#pragma unroll
    for (int e = 0; e < 16; ++e) aP[e] = pl[col * 32 + kmap(e, lane)];

    // --- O(16x96) += P(16x32) * V(32x96) ---
#pragma unroll
    for (int t = 0; t < 6; ++t) {
      v16bf bV;
#pragma unroll
      for (int e = 0; e < 16; ++e)
        bV[e] = V[((size_t)b * kL + k0 + kmap(e, lane)) * kDin + h * kHd + t * 16 + col];
      accO[t] = wmma_bf16(aP, bV, accO[t]);
    }
    __syncthreads();
  }

  const int rb = (lane >> 4) << 3;
#pragma unroll
  for (int r = 0; r < 8; ++r) {
    float inv = 1.f / l8[r];
#pragma unroll
    for (int t = 0; t < 6; ++t)
      O[((size_t)b * kL + q0 + rb + r) * kDin + h * kHd + t * 16 + col] =
          (__bf16)(accO[t][r] * inv);
  }
}

// ---------------------------------------------------------------------------
// Node attention core (d=32, m=48). One wave per (b,h,i); lane = feature dim.
// All activations bf16; fp32 math; masks fp32.
// ---------------------------------------------------------------------------
__global__ void node_core_kernel(const __bf16* __restrict__ q,
                                 const __bf16* __restrict__ ks,
                                 const __bf16* __restrict__ vs,
                                 const __bf16* __restrict__ ws,
                                 const __bf16* __restrict__ kc,
                                 const __bf16* __restrict__ vc,
                                 const __bf16* __restrict__ wc,
                                 const float* __restrict__ mself,
                                 const float* __restrict__ mcross,
                                 int transC, __bf16* __restrict__ out) {
  const int lane = threadIdx.x;
  const int i = blockIdx.x, h = blockIdx.y, b = blockIdx.z;
  __shared__ float sc[2 * kM];
  const float qv = (float)q[((size_t)b * kM + i) * kD + h * 32 + lane];
  const float scale = 0.17677669529663687f; // 1/sqrt(32)

  for (int j = 0; j < kM; ++j) {
    float t = qv * ((float)ks[((size_t)b * kM + j) * kD + h * 32 + lane] +
                    (float)ws[(((size_t)b * kM + i) * kM + j) * kD + h * 32 + lane]);
    for (int off = 16; off >= 1; off >>= 1) t += __shfl_xor(t, off);
    float sv = (mself[b * kM + j] == 0.f) ? -1e9f : t * scale;
    if (lane == 0) sc[j] = sv;
  }
  for (int j = 0; j < kM; ++j) {
    int wi = transC ? j : i, wj = transC ? i : j;
    float t = qv * ((float)kc[((size_t)b * kM + j) * kD + h * 32 + lane] +
                    (float)wc[(((size_t)b * kM + wi) * kM + wj) * kD + h * 32 + lane]);
    for (int off = 16; off >= 1; off >>= 1) t += __shfl_xor(t, off);
    float mv = transC ? mcross[((size_t)b * kM + j) * kM + i]
                      : mcross[((size_t)b * kM + i) * kM + j];
    float sv = (mv == 0.f) ? -1e9f : t * scale;
    if (lane == 0) sc[kM + j] = sv;
  }
  __syncthreads();

  float v0 = sc[lane], v1 = sc[lane + 32], v2 = sc[lane + 64];
  float mx = fmaxf(fmaxf(v0, v1), v2);
  for (int off = 16; off >= 1; off >>= 1) mx = fmaxf(mx, __shfl_xor(mx, off));
  float e0 = __expf(v0 - mx), e1 = __expf(v1 - mx), e2 = __expf(v2 - mx);
  float s = e0 + e1 + e2;
  for (int off = 16; off >= 1; off >>= 1) s += __shfl_xor(s, off);
  float inv = 1.f / s;
  sc[lane] = e0 * inv; sc[lane + 32] = e1 * inv; sc[lane + 64] = e2 * inv;
  __syncthreads();

  float o = 0.f;
  for (int j = 0; j < kM; ++j)
    o += sc[j] * ((float)vs[((size_t)b * kM + j) * kD + h * 32 + lane] +
                  (float)ws[(((size_t)b * kM + i) * kM + j) * kD + h * 32 + lane]);
  for (int j = 0; j < kM; ++j) {
    int wi = transC ? j : i, wj = transC ? i : j;
    o += sc[kM + j] * ((float)vc[((size_t)b * kM + j) * kD + h * 32 + lane] +
                       (float)wc[(((size_t)b * kM + wi) * kM + wj) * kD + h * 32 + lane]);
  }
  out[((size_t)b * kM + i) * kD + h * 32 + lane] = (__bf16)o;
}

// ---------------------------------------------------------------------------
// small utility kernels
// ---------------------------------------------------------------------------
template <typename ET>
__global__ void build_xcat_kernel(const __bf16* __restrict__ fa,
                                  const __bf16* __restrict__ fb,
                                  const ET* __restrict__ edge,
                                  __bf16* __restrict__ X) {
  int idx = blockIdx.x * 256 + threadIdx.x;
  if (idx >= kB * kL * kD) return;
  int c = idx % kD;
  int l = (idx / kD) % kL;
  int b = idx / (kD * kL);
  int i = l / kM, j = l % kM;
  __bf16* xr = X + ((size_t)b * kL + l) * kDin;
  xr[c]          = fa[((size_t)b * kM + i) * kD + c];
  xr[kD + c]     = fb[((size_t)b * kM + j) * kD + c];
  xr[2 * kD + c] = (__bf16)edge[((size_t)b * kL + l) * kD + c];
}

__global__ void build_task_km_kernel(const float* __restrict__ mv,
                                     float* __restrict__ km) {
  int idx = blockIdx.x * 256 + threadIdx.x;
  if (idx >= kB * kL) return;
  int l = idx % kL, b = idx / kL;
  km[idx] = mv[b * kM + l / kM] * mv[b * kM + l % kM];
}

// A bf16, W fp32: used for policy/value Wo heads (Kout = 1 or 4)
__global__ void gemv_cols_bf16A(const __bf16* __restrict__ A,
                                const float* __restrict__ W,
                                float* __restrict__ C, int N, int Kin, int Kout) {
  int idx = blockIdx.x * blockDim.x + threadIdx.x;
  if (idx >= N * Kout) return;
  int row = idx / Kout, c = idx % Kout;
  float s = 0.f;
  const __bf16* a = A + (size_t)row * Kin;
  for (int k = 0; k < Kin; ++k) s += (float)a[k] * W[(size_t)k * Kout + c];
  C[idx] = s;
}

// all-fp32 GEMV with bias: final critic head
__global__ void gemv_cols_f32(const float* __restrict__ A,
                              const float* __restrict__ W,
                              const float* __restrict__ bias,
                              float* __restrict__ C, int N, int Kin, int Kout) {
  int idx = blockIdx.x * blockDim.x + threadIdx.x;
  if (idx >= N * Kout) return;
  int row = idx / Kout, c = idx % Kout;
  float s = bias[c];
  const float* a = A + (size_t)row * Kin;
  for (int k = 0; k < Kin; ++k) s += a[k] * W[(size_t)k * Kout + c];
  C[idx] = s;
}

__global__ void masked_softmax_kernel(const float* __restrict__ act,
                                      const float* __restrict__ mask,
                                      float* __restrict__ out, int n) {
  const int row = blockIdx.x, tid = threadIdx.x;
  __shared__ float red[256];
  float mx = -1e30f;
  for (int i = tid; i < n; i += 256) {
    float v = (mask[(size_t)row * n + i] < 1.f) ? -1e9f : act[(size_t)row * n + i];
    mx = fmaxf(mx, v);
  }
  red[tid] = mx; __syncthreads();
  for (int s = 128; s > 0; s >>= 1) { if (tid < s) red[tid] = fmaxf(red[tid], red[tid + s]); __syncthreads(); }
  mx = red[0]; __syncthreads();
  float sum = 0.f;
  for (int i = tid; i < n; i += 256) {
    float v = (mask[(size_t)row * n + i] < 1.f) ? -1e9f : act[(size_t)row * n + i];
    sum += __expf(v - mx);
  }
  red[tid] = sum; __syncthreads();
  for (int s = 128; s > 0; s >>= 1) { if (tid < s) red[tid] += red[tid + s]; __syncthreads(); }
  sum = red[0]; __syncthreads();
  float inv = 1.f / sum;
  for (int i = tid; i < n; i += 256) {
    float v = (mask[(size_t)row * n + i] < 1.f) ? -1e9f : act[(size_t)row * n + i];
    out[(size_t)row * n + i] = __expf(v - mx) * inv;
  }
}

// ---------------------------------------------------------------------------
// host-side helpers
// ---------------------------------------------------------------------------
struct MhaW { const float *Wq, *Wk, *Wv, *Wo; };

static void launch_gemm(const __bf16* A, const __bf16* WT, __bf16* C,
                        int N, int Kin, int Kout, hipStream_t s) {
  if (N % 64 == 0) {
    dim3 g(Kout / 16, N / 64);
    wmma_gemm_m64<<<g, 32, 0, s>>>(A, WT, C, Kin, Kout);
  } else {
    dim3 g(Kout / 16, N / 16);
    wmma_gemm_n16<<<g, 32, 0, s>>>(A, WT, C, Kin, Kout);
  }
}

// transpose+convert the fp32 weight into WT scratch, then GEMM (stream-serial,
// so a single WT scratch buffer is safe to reuse between launches)
static void gemm_w(const float* W, const __bf16* A, __bf16* C,
                   int N, int Kin, int Kout, __bf16* WT, hipStream_t s) {
  int tot = Kin * Kout;
  transpose_to_bf16_kernel<<<(tot + 255) / 256, 256, 0, s>>>(W, WT, Kin, Kout);
  launch_gemm(A, WT, C, N, Kin, Kout, s);
}

static void run_mha_full(const MhaW& w, const __bf16* X, const float* km,
                         __bf16* Q, __bf16* K, __bf16* V, __bf16* O,
                         __bf16* OUTbf, float* OUTf, int koutWo,
                         __bf16* WT, hipStream_t s) {
  gemm_w(w.Wq, X, Q, kB * kL, kDin, kDin, WT, s);
  gemm_w(w.Wk, X, K, kB * kL, kDin, kDin, WT, s);
  gemm_w(w.Wv, X, V, kB * kL, kDin, kDin, WT, s);
  mha_attn_kernel<<<dim3(kL / 16, kH, kB), 32, 0, s>>>(Q, K, V, km, O);
  if (koutWo % 16 == 0)
    gemm_w(w.Wo, O, OUTbf, kB * kL, kDin, koutWo, WT, s);
  else
    gemv_cols_bf16A<<<(kB * kL * koutWo + 255) / 256, 256, 0, s>>>(
        O, w.Wo, OUTf, kB * kL, kDin, koutWo);
}

static void run_node(const float* const* nw, const __bf16* fs, const __bf16* fo,
                     const __bf16* WS, const __bf16* WC, int transC,
                     const float* mself, const float* cmask,
                     __bf16* qn, __bf16* ksn, __bf16* vsn, __bf16* kcn,
                     __bf16* vcn, __bf16* ON, __bf16* FOUT,
                     __bf16* WT, hipStream_t s) {
  // nw: 0=Wq 1=Wks 2=Wvs 3=Wes 4=Wkc 5=Wvc 6=Wec 7=Wo
  gemm_w(nw[0], fs, qn,  kB * kM, kD, kD, WT, s);
  gemm_w(nw[1], fs, ksn, kB * kM, kD, kD, WT, s);
  gemm_w(nw[2], fs, vsn, kB * kM, kD, kD, WT, s);
  gemm_w(nw[4], fo, kcn, kB * kM, kD, kD, WT, s);
  gemm_w(nw[5], fo, vcn, kB * kM, kD, kD, WT, s);
  node_core_kernel<<<dim3(kM, kH, kB), 32, 0, s>>>(qn, ksn, vsn, WS, kcn, vcn, WC,
                                                   mself, cmask, transC, ON);
  gemm_w(nw[7], ON, FOUT, kB * kM, kD, kD, WT, s);
}

// ---------------------------------------------------------------------------
extern "C" void kernel_launch(void* const* d_in, const int* in_sizes, int n_in,
                              void* d_out, int out_size, void* d_ws, size_t ws_size,
                              hipStream_t stream) {
  (void)in_sizes; (void)n_in; (void)out_size; (void)ws_size;
  auto F = [&](int i) { return (const float*)d_in[i]; };
  const float *f1 = F(0), *f2 = F(1), *e1 = F(2), *e2 = F(3);
  const float *m1v = F(4), *m2v = F(5), *ecost = F(6), *cmask = F(7);
  MhaW coop1{F(8),  F(9),  F(10), F(11)};
  MhaW coop2{F(12), F(13), F(14), F(15)};
  MhaW task1{F(16), F(17), F(18), F(19)};
  MhaW task2{F(20), F(21), F(22), F(23)};
  const float* n1W[8]; for (int i = 0; i < 8; ++i) n1W[i] = F(24 + i);
  const float* n2W[8]; for (int i = 0; i < 8; ++i) n2W[i] = F(32 + i);
  MhaW policy{F(40), F(41), F(42), F(43)};
  MhaW value1{F(44), F(45), F(46), F(47)};
  const float *v2W = F(48), *v2b = F(49);

  // workspace carve (~30 MB; activations bf16)
  char* base = (char*)d_ws;
  size_t off = 0;
  auto takeB  = [&](size_t bytes) { void* p = base + off; off += (bytes + 255) & ~(size_t)255; return p; };
  auto takeBF = [&](size_t n) { return (__bf16*)takeB(n * sizeof(__bf16)); };
  auto takeF  = [&](size_t n) { return (float*)takeB(n * sizeof(float)); };
  const size_t NB = (size_t)kB * kL;
  const size_t NF = (size_t)kB * kM * kD;
  __bf16* XCAT = takeBF(NB * kDin);
  __bf16* Qb = takeBF(NB * kDin);
  __bf16* Kb = takeBF(NB * kDin);
  __bf16* Vb = takeBF(NB * kDin);
  __bf16* Ob = takeBF(NB * kDin);
  __bf16* ECa = takeBF(NB * kD); __bf16* ECb = takeBF(NB * kD);
  __bf16* G1a = takeBF(NB * kD); __bf16* G1b = takeBF(NB * kD);
  __bf16* G2a = takeBF(NB * kD); __bf16* G2b = takeBF(NB * kD);
  __bf16* WS1 = takeBF(NB * kD); __bf16* WS2 = takeBF(NB * kD); __bf16* WCb = takeBF(NB * kD);
  __bf16* F1c = takeBF(NF); __bf16* F2c = takeBF(NF);
  __bf16* F11 = takeBF(NF); __bf16* F21 = takeBF(NF);
  __bf16* F12 = takeBF(NF); __bf16* F22 = takeBF(NF);
  __bf16* qn = takeBF(NF); __bf16* ksn = takeBF(NF); __bf16* vsn = takeBF(NF);
  __bf16* kcn = takeBF(NF); __bf16* vcn = takeBF(NF); __bf16* ON = takeBF(NF);
  __bf16* WT = takeBF((size_t)kDin * kDin);   // weight-transpose scratch
  float* kmt1 = takeF(NB); float* kmt2 = takeF(NB);
  float* ACT = takeF(NB); float* CRIT = takeF(NB * 4);

  const int gKM = (kB * kL + 255) / 256;
  const int gX  = (kB * kL * kD + 255) / 256;
  build_task_km_kernel<<<gKM, 256, 0, stream>>>(m1v, kmt1);
  build_task_km_kernel<<<gKM, 256, 0, stream>>>(m2v, kmt2);
  f32_to_bf16_kernel<<<(int)((NF + 255) / 256), 256, 0, stream>>>(f1, F1c, (int)NF);
  f32_to_bf16_kernel<<<(int)((NF + 255) / 256), 256, 0, stream>>>(f2, F2c, (int)NF);

  // ---- layer 1 ----
  build_xcat_kernel<float><<<gX, 256, 0, stream>>>(F1c, F2c, ecost, XCAT);
  run_mha_full(coop1, XCAT, cmask, Qb, Kb, Vb, Ob, ECa, nullptr, kD, WT, stream);
  build_xcat_kernel<float><<<gX, 256, 0, stream>>>(F1c, F1c, e1, XCAT);
  run_mha_full(task1, XCAT, kmt1, Qb, Kb, Vb, Ob, G1a, nullptr, kD, WT, stream);
  build_xcat_kernel<float><<<gX, 256, 0, stream>>>(F2c, F2c, e2, XCAT);
  run_mha_full(task1, XCAT, kmt2, Qb, Kb, Vb, Ob, G2a, nullptr, kD, WT, stream);

  gemm_w(n1W[3], G1a, WS1, kB * kL, kD, kD, WT, stream);   // Wes
  gemm_w(n1W[3], G2a, WS2, kB * kL, kD, kD, WT, stream);
  gemm_w(n1W[6], ECa, WCb, kB * kL, kD, kD, WT, stream);   // Wec
  run_node(n1W, F1c, F2c, WS1, WCb, 0, m1v, cmask, qn, ksn, vsn, kcn, vcn, ON, F11, WT, stream);
  run_node(n1W, F2c, F1c, WS2, WCb, 1, m2v, cmask, qn, ksn, vsn, kcn, vcn, ON, F21, WT, stream);

  // ---- layer 2 ----
  build_xcat_kernel<__bf16><<<gX, 256, 0, stream>>>(F11, F21, ECa, XCAT);
  run_mha_full(coop2, XCAT, cmask, Qb, Kb, Vb, Ob, ECb, nullptr, kD, WT, stream);
  build_xcat_kernel<__bf16><<<gX, 256, 0, stream>>>(F11, F11, G1a, XCAT);
  run_mha_full(task2, XCAT, kmt1, Qb, Kb, Vb, Ob, G1b, nullptr, kD, WT, stream);
  build_xcat_kernel<__bf16><<<gX, 256, 0, stream>>>(F21, F21, G2a, XCAT);
  run_mha_full(task2, XCAT, kmt2, Qb, Kb, Vb, Ob, G2b, nullptr, kD, WT, stream);

  gemm_w(n2W[3], G1b, WS1, kB * kL, kD, kD, WT, stream);
  gemm_w(n2W[3], G2b, WS2, kB * kL, kD, kD, WT, stream);
  gemm_w(n2W[6], ECb, WCb, kB * kL, kD, kD, WT, stream);
  run_node(n2W, F11, F21, WS1, WCb, 0, m1v, cmask, qn, ksn, vsn, kcn, vcn, ON, F12, WT, stream);
  run_node(n2W, F21, F11, WS2, WCb, 1, m2v, cmask, qn, ksn, vsn, kcn, vcn, ON, F22, WT, stream);

  // ---- heads ----
  build_xcat_kernel<__bf16><<<gX, 256, 0, stream>>>(F12, F22, ECb, XCAT);
  run_mha_full(policy, XCAT, cmask, Qb, Kb, Vb, Ob, nullptr, ACT, 1, WT, stream);
  masked_softmax_kernel<<<kB, 256, 0, stream>>>(ACT, cmask, (float*)d_out, kL);
  run_mha_full(value1, XCAT, cmask, Qb, Kb, Vb, Ob, nullptr, CRIT, 4, WT, stream);
  gemv_cols_f32<<<1, 64, 0, stream>>>(CRIT, v2W, v2b,
                                      (float*)d_out + (size_t)kB * kL,
                                      kB, kM * kM * 4, 16);
}
